// GestureClassifier_21492016349878
// MI455X (gfx1250) — matrix operand order repaired and checked by reading, not compiled, not run on hardware
//
#include <hip/hip_runtime.h>

// ---------------------------------------------------------------------------
// GestureClassifier for MI455X (gfx1250, wave32, WMMA 16x16x32 f16)
//
// Roofline: ~54 GFLOP of MACs vs ~10MB real I/O -> compute bound.
// K = H = 32 exactly matches v_wmma_f32_16x16x32_f16, so every LSTM
// matmul tile is a single WMMA. Each wave owns 16 batch rows end-to-end
// (5 layers x 2 sensors + FC head) -> zero inter-wave synchronization.
// LDS/block = 41.9KB -> ~7 single-wave blocks per 320KB WGP to hide the
// serial recurrence chain.
// ---------------------------------------------------------------------------

typedef __attribute__((ext_vector_type(16))) _Float16 v16h;
typedef __attribute__((ext_vector_type(8)))  float    v8f;

#define HID   32
#define TT    20
#define G4H   128
#define ROWS  16            // batch rows per wave
#define SEQBUF (TT*32*16)   // halfs per ping-pong h-sequence buffer

// d_ws byte offsets (pre-converted f16 WMMA fragments, < 0.5 MB, L2 resident)
#define WS_WHH   0                                  // half [2][5][8][32][16]
#define WS_WIH   (WS_WHH + 2*5*8*32*16*2)           // half [2][4][8][32][16]
#define WS_FC1   (WS_WIH + 2*4*8*32*16*2)           // half [40][8][32][16]
#define WS_BIAS  (WS_FC1 + 40*8*32*16*2)            // float [2][5][128]

__device__ __forceinline__ float sigm(float x)  { return 1.0f / (1.0f + __expf(-x)); }
__device__ __forceinline__ float tanhp(float x) { return 2.0f / (1.0f + __expf(-2.0f * x)) - 1.0f; }

// ---------------------------------------------------------------------------
// Prep kernels: convert f32 weights into f16 WMMA B-fragment layout.
// B-fragment (32x16, f16): lane n (n&15 = column), halves jj -> K = (lane>>4)*16+jj
// B[k][n] = W[nglobal][k]  (since gates = x @ W^T)
// ---------------------------------------------------------------------------
__global__ void k_prep_whh(const float* __restrict__ aW, const float* __restrict__ gW,
                           _Float16* __restrict__ o) {
    int idx = blockIdx.x * blockDim.x + threadIdx.x;
    if (idx >= 2*5*8*32*16) return;
    int jj = idx & 15, lane = (idx >> 4) & 31, j = (idx >> 9) & 7;
    int sl = idx >> 12, layer = sl % 5, s = sl / 5;
    const float* W = s ? gW : aW;                    // [5][128][32]
    int row = j*16 + (lane & 15);
    int k   = (lane >> 4)*16 + jj;
    o[idx] = (_Float16)W[(layer*G4H + row)*HID + k];
}

__global__ void k_prep_wih(const float* __restrict__ aW, const float* __restrict__ gW,
                           _Float16* __restrict__ o) {
    int idx = blockIdx.x * blockDim.x + threadIdx.x;
    if (idx >= 2*4*8*32*16) return;
    int jj = idx & 15, lane = (idx >> 4) & 31, j = (idx >> 9) & 7;
    int sl = idx >> 12, layer = sl % 4, s = sl / 4;
    const float* W = s ? gW : aW;                    // [4][128][32]
    int row = j*16 + (lane & 15);
    int k   = (lane >> 4)*16 + jj;
    o[idx] = (_Float16)W[(layer*G4H + row)*HID + k];
}

__global__ void k_prep_fc1(const float* __restrict__ W, _Float16* __restrict__ o) {
    int idx = blockIdx.x * blockDim.x + threadIdx.x;
    if (idx >= 40*8*32*16) return;
    int jj = idx & 15, lane = (idx >> 4) & 31, jn = (idx >> 9) & 7;
    int kc = idx >> 12;                              // K-chunk 0..39 (seq position)
    int row = jn*16 + (lane & 15);                   // fc1 output col 0..127
    int k   = kc*HID + (lane >> 4)*16 + jj;          // feature 0..1279
    o[idx] = (_Float16)W[row*(2*TT*HID) + k];        // fc1_w [128][1280]
}

__global__ void k_prep_bias(const float* __restrict__ a_bih, const float* __restrict__ a_bhh,
                            const float* __restrict__ g_bih, const float* __restrict__ g_bhh,
                            float* __restrict__ o) {
    int idx = blockIdx.x * blockDim.x + threadIdx.x;
    if (idx >= 2*5*G4H) return;
    int n = idx % G4H, sl = idx / G4H, layer = sl % 5, s = sl / 5;
    const float* bi = s ? g_bih : a_bih;
    const float* bh = s ? g_bhh : a_bhh;
    o[idx] = bi[layer*G4H + n] + bh[layer*G4H + n];
}

// ---------------------------------------------------------------------------
// Pointwise LSTM cell + repack of h into the 16x32 f16 A-fragment layout.
// Accumulator layout (16x16 f32): lane l holds col n=(l&15), rows m=r+8*(l>>4).
// A-fragment (16x32 f16): lane l holds row m=(l&15);
//   l<16 : halves 0..15 = K0..15 (contiguous, matching the B prep)
//   l>=16: halves 0..15 = K16..31
// Bounce through per-wave LDS (same-wave LDS ops are in-order; s_wait_dscnt
// covers data availability).
// ---------------------------------------------------------------------------
__device__ __forceinline__ v16h cell_update(v8f* acc, v8f& c0, v8f& c1,
                                            _Float16* __restrict__ xp,
                                            int lane, _Float16* __restrict__ seq_dst) {
    const int ln15 = lane & 15, lhi = lane >> 4;
    #pragma unroll
    for (int hc = 0; hc < 2; ++hc) {
        v8f& c = hc ? c1 : c0;
        #pragma unroll
        for (int r = 0; r < 8; ++r) {
            float iv = sigm (acc[0 + hc][r]);
            float fv = sigm (acc[2 + hc][r]);
            float gv = tanhp(acc[4 + hc][r]);
            float ov = sigm (acc[6 + hc][r]);
            float cv = fv * c[r] + iv * gv;
            c[r] = cv;
            float hv = ov * tanhp(cv);
            xp[(r + 8*lhi)*HID + hc*16 + ln15] = (_Float16)hv;
        }
    }
    asm volatile("s_wait_dscnt 0" ::: "memory");
    const int kb = lhi * 16;
    v16h hf;
    #pragma unroll
    for (int j = 0; j < 8; ++j) hf[j]     = xp[ln15*HID + kb + j];
    #pragma unroll
    for (int j = 0; j < 8; ++j) hf[8 + j] = xp[ln15*HID + kb + 8 + j];
    *(v16h*)seq_dst = hf;   // next layer's x_t A-fragment, contiguous 32B/lane
    return hf;
}

// ---------------------------------------------------------------------------
// Main fused kernel: one wave = 16 batch rows, full pipeline.
// ---------------------------------------------------------------------------
__global__ __launch_bounds__(32)
void lstm_fused(const float* __restrict__ accel, const float* __restrict__ gyro,
                const float* __restrict__ a_wih0, const float* __restrict__ g_wih0,
                const float* __restrict__ fc1_b,  const float* __restrict__ fc2_w,
                const float* __restrict__ fc2_b,
                const _Float16* __restrict__ whh_frag_g,
                const _Float16* __restrict__ wih_frag_g,
                const _Float16* __restrict__ fc1_frag_g,
                const float* __restrict__ bias_g,
                float* __restrict__ out) {
    __shared__ __align__(32) _Float16 s_seq[2][TT][32][16];   // h-seq ping-pong (A-frag layout)
    __shared__ __align__(32) _Float16 s_xp[ROWS][HID];        // transpose bounce

    const int lane = threadIdx.x & 31;
    const int ln15 = lane & 15;
    const int lhi  = lane >> 4;
    const long batch_base = (long)blockIdx.x * ROWS;

    // NOTE: no local array of LDS-derived pointers (ld.lld cannot relocate an
    // addrspacecast in a static initializer) -- use base + runtime offset.
    _Float16* seqbase = &s_seq[0][0][0][0];
    _Float16* xp      = &s_xp[0][0];

    // fc1 accumulators, bias preloaded, live across both sensor streams
    v8f fc_acc[8];
    #pragma unroll
    for (int jn = 0; jn < 8; ++jn) {
        float bv = fc1_b[jn*16 + ln15];
        #pragma unroll
        for (int r = 0; r < 8; ++r) fc_acc[jn][r] = bv;
    }

    for (int s = 0; s < 2; ++s) {
        const float* x0  = s ? gyro   : accel;
        const float* w0p = s ? g_wih0 : a_wih0;

        // ---------------- layer 0 (D_in = 3, VALU input proj) ----------------
        v16h whhf[8];
        #pragma unroll
        for (int j = 0; j < 8; ++j)
            whhf[j] = *(const v16h*)(whh_frag_g + ((((s*5 + 0)*8 + j)*32 + lane) << 4));
        float b0[8], w00[8], w01[8], w02[8];
        #pragma unroll
        for (int j = 0; j < 8; ++j) {
            int n = j*16 + ln15;
            b0[j]  = bias_g[(s*5 + 0)*G4H + n];
            w00[j] = w0p[n*3 + 0];
            w01[j] = w0p[n*3 + 1];
            w02[j] = w0p[n*3 + 2];
        }
        {
            v16h hfrag = {};
            v8f  c0 = {}, c1 = {};
            for (int t = 0; t < TT; ++t) {
                v8f acc[8];
                #pragma unroll
                for (int r = 0; r < 8; ++r) {
                    long grow = batch_base + r + 8*lhi;
                    const float* xr = x0 + grow*(TT*3) + t*3;
                    float d0 = xr[0], d1 = xr[1], d2 = xr[2];
                    #pragma unroll
                    for (int j = 0; j < 8; ++j)
                        acc[j][r] = b0[j] + d0*w00[j] + d1*w01[j] + d2*w02[j];
                }
                #pragma unroll
                for (int j = 0; j < 8; ++j)
                    acc[j] = __builtin_amdgcn_wmma_f32_16x16x32_f16(
                        false, hfrag, false, whhf[j], (short)0, acc[j], false, false);
                hfrag = cell_update(acc, c0, c1, xp, lane, seqbase + t*512 + lane*16);
            }
        }

        // ---------------- layers 1..4 (full WMMA) ----------------
        int cur = 0;
        for (int li = 1; li < 5; ++li) {
            v16h wihf[8];
            #pragma unroll
            for (int j = 0; j < 8; ++j) {
                wihf[j] = *(const v16h*)(wih_frag_g + ((((s*4 + (li-1))*8 + j)*32 + lane) << 4));
                whhf[j] = *(const v16h*)(whh_frag_g + ((((s*5 + li    )*8 + j)*32 + lane) << 4));
            }
            float bl[8];
            #pragma unroll
            for (int j = 0; j < 8; ++j) bl[j] = bias_g[(s*5 + li)*G4H + j*16 + ln15];

            _Float16* curb = seqbase + cur * SEQBUF;
            _Float16* nxtb = seqbase + (cur ^ 1) * SEQBUF;
            v16h hfrag = {};
            v8f  c0 = {}, c1 = {};
            for (int t = 0; t < TT; ++t) {
                v16h xfrag = *(const v16h*)(curb + t*512 + lane*16);
                v8f acc[8];
                #pragma unroll
                for (int j = 0; j < 8; ++j) {
                    v8f a;
                    #pragma unroll
                    for (int r = 0; r < 8; ++r) a[r] = bl[j];
                    a = __builtin_amdgcn_wmma_f32_16x16x32_f16(
                            false, xfrag, false, wihf[j], (short)0, a, false, false);
                    a = __builtin_amdgcn_wmma_f32_16x16x32_f16(
                            false, hfrag, false, whhf[j], (short)0, a, false, false);
                    acc[j] = a;
                }
                hfrag = cell_update(acc, c0, c1, xp, lane, nxtb + t*512 + lane*16);
            }
            cur ^= 1;
        }
        // final layer output sequence lives in buffer `cur` (== 0 after 4 flips)

        // ---------------- fold this sensor into FC1 accumulators ----------------
        _Float16* fin = seqbase + cur * SEQBUF;
        for (int t = 0; t < TT; ++t) {
            v16h a = *(const v16h*)(fin + t*512 + lane*16);
            int kc = s*TT + t;   // feature chunk in concat([accel_seq, gyro_seq])
            #pragma unroll
            for (int jn = 0; jn < 8; ++jn) {
                v16h bfr = *(const v16h*)(fc1_frag_g + (((kc*8 + jn)*32 + lane) << 4));
                fc_acc[jn] = __builtin_amdgcn_wmma_f32_16x16x32_f16(
                                 false, a, false, bfr, (short)0, fc_acc[jn], false, false);
            }
        }
    }

    // ---------------- ReLU + FC2 (N=5, VALU via LDS) ----------------
    // Reuse seq buffer 1 as the fc1-output scratch: after the second sensor the
    // final sequence is in buffer 0 and buffer 1 is dead (8KB needed <= 20KB).
    float* fcs = (float*)(seqbase + SEQBUF);
    #pragma unroll
    for (int jn = 0; jn < 8; ++jn)
        #pragma unroll
        for (int r = 0; r < 8; ++r) {
            float v = fc_acc[jn][r];
            v = v > 0.0f ? v : 0.0f;
            fcs[(r + 8*lhi)*G4H + jn*16 + ln15] = v;
        }
    asm volatile("s_wait_dscnt 0" ::: "memory");

    for (int idx = lane; idx < ROWS*5; idx += 32) {
        int m = idx / 5, q = idx - m*5;
        float sum = fc2_b[q];
        #pragma unroll 4
        for (int n = 0; n < G4H; ++n) sum += fcs[m*G4H + n] * fc2_w[q*G4H + n];
        out[(batch_base + m)*5 + q] = sum;
    }
}

// ---------------------------------------------------------------------------
extern "C" void kernel_launch(void* const* d_in, const int* in_sizes, int n_in,
                              void* d_out, int out_size, void* d_ws, size_t ws_size,
                              hipStream_t stream) {
    const float* accel      = (const float*)d_in[0];
    const float* gyro       = (const float*)d_in[1];
    const float* a_wih0     = (const float*)d_in[2];
    const float* a_wih_rest = (const float*)d_in[3];
    const float* a_whh      = (const float*)d_in[4];
    const float* a_bih      = (const float*)d_in[5];
    const float* a_bhh      = (const float*)d_in[6];
    const float* g_wih0     = (const float*)d_in[7];
    const float* g_wih_rest = (const float*)d_in[8];
    const float* g_whh      = (const float*)d_in[9];
    const float* g_bih      = (const float*)d_in[10];
    const float* g_bhh      = (const float*)d_in[11];
    const float* fc1_w      = (const float*)d_in[12];
    const float* fc1_b      = (const float*)d_in[13];
    const float* fc2_w      = (const float*)d_in[14];
    const float* fc2_b      = (const float*)d_in[15];
    float* out = (float*)d_out;

    _Float16* ws_whh  = (_Float16*)((char*)d_ws + WS_WHH);
    _Float16* ws_wih  = (_Float16*)((char*)d_ws + WS_WIH);
    _Float16* ws_fc1  = (_Float16*)((char*)d_ws + WS_FC1);
    float*    ws_bias = (float*)   ((char*)d_ws + WS_BIAS);

    const int B = in_sizes[0] / (TT * 3);

    k_prep_whh <<<(2*5*8*32*16 + 255)/256, 256, 0, stream>>>(a_whh, g_whh, ws_whh);
    k_prep_wih <<<(2*4*8*32*16 + 255)/256, 256, 0, stream>>>(a_wih_rest, g_wih_rest, ws_wih);
    k_prep_fc1 <<<(40*8*32*16  + 255)/256, 256, 0, stream>>>(fc1_w, ws_fc1);
    k_prep_bias<<<(2*5*G4H     + 255)/256, 256, 0, stream>>>(a_bih, a_bhh, g_bih, g_bhh, ws_bias);

    const int blocks = B / ROWS;   // one wave (32 threads) per 16 batch rows
    lstm_fused<<<blocks, 32, 0, stream>>>(accel, gyro, a_wih0, g_wih0,
                                          fc1_b, fc2_w, fc2_b,
                                          ws_whh, ws_wih, ws_fc1, ws_bias, out);
    (void)n_in; (void)out_size; (void)ws_size;
}
